// SlidingWindowAttention_73315091743311
// MI455X (gfx1250) — compile-verified, matrix operand order checked by reading, not run on hardware
//
#include <hip/hip_runtime.h>
#include <hip/hip_bf16.h>

typedef __attribute__((ext_vector_type(16))) _Float16 v16h;
typedef __attribute__((ext_vector_type(8)))  _Float16 v8h;
typedef __attribute__((ext_vector_type(8)))  float    v8f;

#define NTOK   16384
#define DMODEL 512
#define NHEAD  8
#define HD     64
#define BLK    256                 // query block == w_half
#define WH     256                 // w_half
#define MPAD   (NTOK + 2*WH)       // 16896 padded rows
#define NBLK   (NTOK / BLK)        // 64

static __device__ inline v16h frag16(const _Float16* p) {
  // CDNA5 16-bit A/B fragment: per-lane = two contiguous 8-half chunks
  // (caller pre-applies off = (lane>>4)*8; chunks at +0 and +16)
  v8h lo = *(const v8h*)(p);
  v8h hi = *(const v8h*)(p + 16);
  return __builtin_shufflevector(lo, hi, 0,1,2,3,4,5,6,7,8,9,10,11,12,13,14,15);
}

static __device__ inline v8f vzero8() {
  v8f z = {0.f,0.f,0.f,0.f,0.f,0.f,0.f,0.f};
  return z;
}

static __device__ inline float swapx16(float x) {
  // ds_swizzle SWAPX16 (xor 0x10): combine lane l <-> l+16 within wave32
  return __int_as_float(__builtin_amdgcn_ds_swizzle(__float_as_int(x), 0x401f));
}

// ---------------------------------------------------------------- cast f32->f16
__global__ void cast_w_kernel(const float* __restrict__ src,
                              _Float16* __restrict__ dst, int n) {
  int i = (blockIdx.x * blockDim.x + threadIdx.x) * 4;
  if (i + 3 < n) {
    float4 v = *(const float4*)(src + i);
    dst[i+0] = (_Float16)v.x; dst[i+1] = (_Float16)v.y;
    dst[i+2] = (_Float16)v.z; dst[i+3] = (_Float16)v.w;
  }
}

// ------------------------------------------------- gather by phi + circular pad
__global__ void gather_pad_kernel(const float* __restrict__ x,
                                  const int* __restrict__ phi,
                                  _Float16* __restrict__ xp) {
  int t   = blockIdx.x * 256 + threadIdx.x;   // one 8-half segment per thread
  int idx = t * 8;
  int p   = idx >> 9;          // padded row
  int c   = idx & 511;         // column
  int o   = p - WH;
  if (o < 0)      o += NTOK;
  if (o >= NTOK)  o -= NTOK;
  int src = phi[o];
  const float* s = x + (size_t)src * DMODEL + c;
  float4 a = *(const float4*)s;
  float4 b = *(const float4*)(s + 4);
  v8h out;
  out[0]=(_Float16)a.x; out[1]=(_Float16)a.y; out[2]=(_Float16)a.z; out[3]=(_Float16)a.w;
  out[4]=(_Float16)b.x; out[5]=(_Float16)b.y; out[6]=(_Float16)b.z; out[7]=(_Float16)b.w;
  *(v8h*)(xp + (size_t)p * DMODEL + c) = out;
}

// ------------------------------------------------------- C = A @ W^T + bias
// A: [M x 512] f16 row-major.  W: [512 x 512] f16 row-major (B(k,n)=W[n][k]).
// 128x128 tile per workgroup, 8 waves (4 along M x 2 along N), 32-deep k-steps.
// Double-buffered LDS: global stage of step kb+1 overlaps WMMA of step kb;
// one barrier per k-step.
template<bool HALF_OUT>
__global__ __launch_bounds__(256)
void gemm512_kernel(const _Float16* __restrict__ A,
                    const _Float16* __restrict__ W,
                    const float* __restrict__ bias,
                    float* __restrict__ outF,
                    _Float16* __restrict__ outH) {
  __shared__ _Float16 As[2][128 * 32];
  __shared__ _Float16 Bs[2][128 * 32];

  const int tid  = threadIdx.x;
  const int lane = tid & 31;
  const int wave = tid >> 5;
  const int m0 = blockIdx.x * 128;
  const int n0 = blockIdx.y * 128;
  const int mwave = (wave & 3) * 32;   // 4 waves along M
  const int nwave = (wave >> 2) * 64;  // 2 waves along N
  const int half = lane >> 4;
  const int l16  = lane & 15;
  const int off  = half * 8;

  const int lrow = tid >> 2;           // 0..63
  const int lseg = (tid & 3) * 8;      // 0,8,16,24

  auto stage = [&](int kb, int buf) {
    const int k0 = kb * 32;
    *(v8h*)&As[buf][ lrow     *32 + lseg] = *(const v8h*)&A[(size_t)(m0+lrow   )*512 + k0 + lseg];
    *(v8h*)&As[buf][(lrow+64) *32 + lseg] = *(const v8h*)&A[(size_t)(m0+lrow+64)*512 + k0 + lseg];
    *(v8h*)&Bs[buf][ lrow     *32 + lseg] = *(const v8h*)&W[(size_t)(n0+lrow   )*512 + k0 + lseg];
    *(v8h*)&Bs[buf][(lrow+64) *32 + lseg] = *(const v8h*)&W[(size_t)(n0+lrow+64)*512 + k0 + lseg];
  };

  v8f acc[2][4];
#pragma unroll
  for (int mt = 0; mt < 2; ++mt)
#pragma unroll
    for (int nt = 0; nt < 4; ++nt) acc[mt][nt] = vzero8();

  stage(0, 0);
  __syncthreads();

  for (int kb = 0; kb < 16; ++kb) {
    const int cur = kb & 1;
    if (kb < 15) stage(kb + 1, cur ^ 1);   // overlap global stage with compute

    v16h aF[2], bF[4];
#pragma unroll
    for (int mt = 0; mt < 2; ++mt)
      aF[mt] = frag16(&As[cur][(mwave + mt*16 + l16) * 32 + off]);
#pragma unroll
    for (int nt = 0; nt < 4; ++nt)
      bF[nt] = frag16(&Bs[cur][(nwave + nt*16 + l16) * 32 + off]);

#pragma unroll
    for (int mt = 0; mt < 2; ++mt)
#pragma unroll
      for (int nt = 0; nt < 4; ++nt)
        acc[mt][nt] = __builtin_amdgcn_wmma_f32_16x16x32_f16(
            false, aF[mt], false, bF[nt], (short)0, acc[mt][nt], false, false);
    __syncthreads();
  }

#pragma unroll
  for (int mt = 0; mt < 2; ++mt) {
#pragma unroll
    for (int nt = 0; nt < 4; ++nt) {
      const int n = n0 + nwave + nt*16 + l16;
      const float bb = bias[n];
      const int mbase = m0 + mwave + mt*16 + half*8;
#pragma unroll
      for (int r = 0; r < 8; ++r) {
        const float val = acc[mt][nt][r] + bb;
        const size_t oi = (size_t)(mbase + r) * 512 + n;
        if (HALF_OUT) outH[oi] = (_Float16)val;
        else          outF[oi] = val;
      }
    }
  }
}

// -------------------------------------------- blocked local attention (flash)
// One workgroup per (block b, head h). 8 waves x 32 queries = 256 queries.
// Computes S^T = K.Q^T so softmax stats are per-lane and exp(S^T) tiles feed
// the O^T = V^T.P^T WMMA directly (no cross-lane shuffle of probabilities).
// Double-buffered K/V staging; wave-uniform band classification per 32-key
// chunk: skip fully-invalid chunks, mask only boundary chunks.
__global__ __launch_bounds__(256)
void attn_kernel(const _Float16* __restrict__ qh,
                 const _Float16* __restrict__ kh,
                 const _Float16* __restrict__ vh,
                 const int* __restrict__ phi,
                 _Float16* __restrict__ outu) {
  __shared__ _Float16 Ks[2][32 * 64];    // [key][d] chunk
  __shared__ _Float16 VsT[2][64 * 32];   // [d][key] chunk (transposed)

  const int b = blockIdx.x, h = blockIdx.y;
  const int tid  = threadIdx.x;
  const int lane = tid & 31;
  const int wave = tid >> 5;
  const int half = lane >> 4;
  const int l16  = lane & 15;
  const int off  = half * 8;
  const int q0   = wave * 32;

  // Q^T fragments (B operand): col q = l16, contiguous-in-d chunks per lane
  v16h qF[2][2];
#pragma unroll
  for (int qt = 0; qt < 2; ++qt) {
    const size_t qrow = (size_t)(b*BLK + WH + q0 + qt*16 + l16);
    const _Float16* qb = qh + qrow * 512 + h * HD;
#pragma unroll
    for (int ds = 0; ds < 2; ++ds)
      qF[qt][ds] = frag16(qb + ds*32 + off);
  }

  float mrun[2] = {-1e30f, -1e30f};
  float lrun[2] = {0.f, 0.f};
  v8f oacc[2][4];
#pragma unroll
  for (int qt = 0; qt < 2; ++qt)
#pragma unroll
    for (int dt = 0; dt < 4; ++dt) oacc[qt][dt] = vzero8();

  const int lr = tid >> 3;        // staged row 0..31
  const int ls = (tid & 7) * 8;   // seg 0..56

  auto stage = [&](int c, int buf) {
    const size_t kb = (size_t)(b * BLK + c * 32);
    *(v8h*)&Ks[buf][lr * 64 + ls] = *(const v8h*)&kh[(kb + lr) * 512 + h * HD + ls];
    v8h vv = *(const v8h*)&vh[(kb + lr) * 512 + h * HD + ls];
#pragma unroll
    for (int i = 0; i < 8; ++i)
      VsT[buf][(ls + i) * 32 + lr] = vv[i];
  };

  stage(0, 0);
  __syncthreads();

  for (int c = 0; c < 24; ++c) {              // 768-key window in 32-key chunks
    const int cur = c & 1;
    if (c < 23) stage(c + 1, cur ^ 1);        // overlap with compute below

    // wave-uniform band classification for this 32-key chunk
    const int c32 = c * 32;
    const bool anyv  = (c32 + 31 >= q0) && (c32 <= q0 + 31 + 2*WH);
    const bool fullv = (c32 >= q0 + 31) && (c32 + 31 <= q0 + 2*WH);

    if (anyv) {
      // K fragments (A operand of S^T): row = key, chunks along d
      v16h aK[2][2];
#pragma unroll
      for (int kt = 0; kt < 2; ++kt)
#pragma unroll
        for (int ds = 0; ds < 2; ++ds)
          aK[kt][ds] = frag16(&Ks[cur][(kt*16 + l16) * 64 + ds*32 + off]);

      v16h pF[2];
#pragma unroll
      for (int qt = 0; qt < 2; ++qt) {
        v8f st[2];
#pragma unroll
        for (int kt = 0; kt < 2; ++kt) {
          v8f s = vzero8();
          s = __builtin_amdgcn_wmma_f32_16x16x32_f16(false, aK[kt][0], false, qF[qt][0], (short)0, s, false, false);
          s = __builtin_amdgcn_wmma_f32_16x16x32_f16(false, aK[kt][1], false, qF[qt][1], (short)0, s, false, false);
          st[kt] = s;
        }
        const int j = q0 + qt*16 + l16;        // query index in block
        float cmax = -3e30f;
        if (fullv) {
#pragma unroll
          for (int kt = 0; kt < 2; ++kt)
#pragma unroll
            for (int r = 0; r < 8; ++r) {
              const float sv = st[kt][r] * 0.125f;
              st[kt][r] = sv;
              cmax = fmaxf(cmax, sv);
            }
        } else {
#pragma unroll
          for (int kt = 0; kt < 2; ++kt)
#pragma unroll
            for (int r = 0; r < 8; ++r) {
              const int rr = c32 + kt*16 + half*8 + r;   // key index in window
              // band mask j <= rr <= j+512 as one unsigned range test
              const float sv = ((unsigned)(rr - j) <= (unsigned)(2*WH))
                                 ? st[kt][r] * 0.125f : -2e30f;
              st[kt][r] = sv;
              cmax = fmaxf(cmax, sv);
            }
        }
        cmax = fmaxf(cmax, swapx16(cmax));
        const float mnew = fmaxf(mrun[qt], cmax);
        const float corr = __expf(mrun[qt] - mnew);
        float p0[8], p1[8];
        float rs = 0.f;
#pragma unroll
        for (int r = 0; r < 8; ++r) {
          p0[r] = __expf(st[0][r] - mnew);
          p1[r] = __expf(st[1][r] - mnew);
          rs += p0[r] + p1[r];
        }
        rs += swapx16(rs);
        lrun[qt] = lrun[qt] * corr + rs;
        mrun[qt] = mnew;
#pragma unroll
        for (int dt = 0; dt < 4; ++dt)
#pragma unroll
          for (int r = 0; r < 8; ++r) oacc[qt][dt][r] *= corr;
        // P^T B-fragment: lo chunk = keytile 0 rows, hi chunk = keytile 1 rows
        v16h p;
#pragma unroll
        for (int r = 0; r < 8; ++r) {
          p[r]     = (_Float16)p0[r];
          p[r + 8] = (_Float16)p1[r];
        }
        pF[qt] = p;
      }

      // O^T += V^T . P^T
#pragma unroll
      for (int dt = 0; dt < 4; ++dt) {
        v16h vF = frag16(&VsT[cur][(dt*16 + l16) * 32 + off]);
#pragma unroll
        for (int qt = 0; qt < 2; ++qt)
          oacc[qt][dt] = __builtin_amdgcn_wmma_f32_16x16x32_f16(
              false, vF, false, pF[qt], (short)0, oacc[qt][dt], false, false);
      }
    }
    __syncthreads();
  }

  // normalize + scatter (out_un[phi[o]] = out[o]) as f16, 16B stores
#pragma unroll
  for (int qt = 0; qt < 2; ++qt) {
    const int j    = q0 + qt*16 + l16;
    const int dest = phi[b * BLK + j];
    const float inv = 1.0f / lrun[qt];
#pragma unroll
    for (int dt = 0; dt < 4; ++dt) {
      v8h o;
#pragma unroll
      for (int r = 0; r < 8; ++r) o[r] = (_Float16)(oacc[qt][dt][r] * inv);
      *(v8h*)&outu[(size_t)dest * 512 + h * HD + dt*16 + half*8] = o;
    }
  }
}

// ---------------------------------------------------------------------------
extern "C" void kernel_launch(void* const* d_in, const int* in_sizes, int n_in,
                              void* d_out, int out_size, void* d_ws, size_t ws_size,
                              hipStream_t stream) {
  const float* x   = (const float*)d_in[0];
  const int*   phi = (const int*)  d_in[1];
  const float* Wq  = (const float*)d_in[2];
  const float* bq  = (const float*)d_in[3];
  const float* Wk  = (const float*)d_in[4];
  const float* bk  = (const float*)d_in[5];
  const float* Wv  = (const float*)d_in[6];
  const float* bv  = (const float*)d_in[7];
  const float* Wo  = (const float*)d_in[8];
  const float* bo  = (const float*)d_in[9];
  float* out = (float*)d_out;

  _Float16* ws = (_Float16*)d_ws;
  size_t o = 0;
  _Float16* xp  = ws + o; o += (size_t)MPAD * DMODEL;
  _Float16* wqh = ws + o; o += (size_t)DMODEL * DMODEL;
  _Float16* wkh = ws + o; o += (size_t)DMODEL * DMODEL;
  _Float16* wvh = ws + o; o += (size_t)DMODEL * DMODEL;
  _Float16* woh = ws + o; o += (size_t)DMODEL * DMODEL;
  _Float16* q_h = ws + o; o += (size_t)MPAD * DMODEL;
  _Float16* k_h = ws + o; o += (size_t)MPAD * DMODEL;
  _Float16* v_h = ws + o; o += (size_t)MPAD * DMODEL;
  _Float16* att = ws + o; o += (size_t)NTOK * DMODEL;

  const int WN = DMODEL * DMODEL;
  cast_w_kernel<<<256, 256, 0, stream>>>(Wq, wqh, WN);
  cast_w_kernel<<<256, 256, 0, stream>>>(Wk, wkh, WN);
  cast_w_kernel<<<256, 256, 0, stream>>>(Wv, wvh, WN);
  cast_w_kernel<<<256, 256, 0, stream>>>(Wo, woh, WN);

  gather_pad_kernel<<<MPAD / 4, 256, 0, stream>>>(x, phi, xp);

  gemm512_kernel<true ><<<dim3(MPAD / 128, 4), 256, 0, stream>>>(xp, wqh, bq, nullptr, q_h);
  gemm512_kernel<true ><<<dim3(MPAD / 128, 4), 256, 0, stream>>>(xp, wkh, bk, nullptr, k_h);
  gemm512_kernel<true ><<<dim3(MPAD / 128, 4), 256, 0, stream>>>(xp, wvh, bv, nullptr, v_h);

  attn_kernel<<<dim3(NBLK, NHEAD), 256, 0, stream>>>(q_h, k_h, v_h, phi, att);

  gemm512_kernel<false><<<dim3(NTOK / 128, 4), 256, 0, stream>>>(att, woh, bo, out, nullptr);
}
